// Decoder_6485400617611
// MI455X (gfx1250) — compile-verified, matrix operand order checked by reading, not run on hardware
//
#include <hip/hip_runtime.h>
#include <hip/hip_bf16.h>
#include <math.h>

// ---------------- types ----------------
typedef __attribute__((ext_vector_type(16))) _Float16 v16h;
typedef __attribute__((ext_vector_type(8)))  _Float16 v8h;
typedef __attribute__((ext_vector_type(8)))  float    v8f;

// Problem constants (match reference)
#define BB 32
#define TT 64
#define SS 64
#define VV 32000
#define EE 512
#define HH 512
#define SDD 1024
#define KCAT 1536      // E + H (inp) + H (hidden) = 512+512+512
#define NG   2048      // 4*H gates

// -------------------------------------------------------------------------
// WMMA fragment loader: 16x16x32 f16 (ISA 7.12.2 striping).
// lanes 0-15 -> row/col = lane, halves[0..7]=K 0..7, [8..15]=K 16..23
// lanes 16-31-> row/col = lane-16, halves[0..7]=K 8..15, [8..15]=K 24..31
// Both A (row-major, K contiguous) and B (N-major rows of K) use this.
// -------------------------------------------------------------------------
__device__ __forceinline__ v16h load_frag16(const _Float16* __restrict__ rowk,
                                            unsigned lane) {
  const int lo = (lane & 16) ? 8 : 0;
  union { v16h v; v8h h[2]; } u;
  u.h[0] = *(const v8h*)(rowk + lo);
  u.h[1] = *(const v8h*)(rowk + lo + 16);
  return u.v;
}

__device__ __forceinline__ v8f wmma_f16(v16h a, v16h b, v8f c) {
  return __builtin_amdgcn_wmma_f32_16x16x32_f16(false, a, false, b,
                                                (short)0, c, false, false);
}

// ---------------- setup kernels ----------------
__global__ void cvt_f16_kernel(const float* __restrict__ s,
                               _Float16* __restrict__ d, int n) {
  int i = blockIdx.x * 256 + threadIdx.x;
  if (i < n) d[i] = (_Float16)s[i];
}

// Wcat[n][k]: k<1024 -> W_ih[n][k] ; k>=1024 -> W_hh[n][k-1024]   (2048 x 1536)
__global__ void build_wcat_kernel(const float* __restrict__ W_ih,
                                  const float* __restrict__ W_hh,
                                  _Float16* __restrict__ wcat) {
  int i = blockIdx.x * 256 + threadIdx.x;
  if (i >= NG * KCAT) return;
  int k = i % KCAT, n = i / KCAT;
  float v = (k < 1024) ? W_ih[(size_t)n * 1024 + k]
                       : W_hh[(size_t)n * 512 + (k - 1024)];
  wcat[i] = (_Float16)v;
}

// x16[(t*32+b)*512 + e] = emb[trg[b][t]][e]
__global__ void embed_kernel(const float* __restrict__ emb,
                             const int* __restrict__ trg,
                             _Float16* __restrict__ x16) {
  int i = blockIdx.x * 256 + threadIdx.x;
  if (i >= TT * BB * EE) return;
  int e = i & 511, bt = i >> 9;
  int b = bt & 31, t = bt >> 5;
  int tok = trg[b * TT + t];
  x16[i] = (_Float16)emb[(size_t)tok * EE + e];
}

__global__ void init_state_kernel(const float* __restrict__ hidden_init,
                                  const float* __restrict__ input_feed_init,
                                  float* __restrict__ h, float* __restrict__ c,
                                  float* __restrict__ hhatf,
                                  _Float16* __restrict__ h16,
                                  _Float16* __restrict__ hhat16) {
  int i = blockIdx.x * 256 + threadIdx.x;
  if (i >= BB * HH) return;
  int j = i & 511;
  float hv = hidden_init[j], cv = hidden_init[HH + j], fv = input_feed_init[j];
  h[i] = hv; c[i] = cv; hhatf[i] = fv;
  h16[i] = (_Float16)hv; hhat16[i] = (_Float16)fv;
}

// ---------------- recurrent-step kernels ----------------
// gates = [x_t, hhat, h] @ Wcat^T  (M=32, N=2048, K=1536).
// 2 N-tiles per wave; K loop split per segment so A loads are GLOBAL (not flat).
__global__ void gate_gemm_kernel(const _Float16* __restrict__ x16_t,
                                 const _Float16* __restrict__ hhat16,
                                 const _Float16* __restrict__ h16,
                                 const _Float16* __restrict__ wcat16,
                                 float* __restrict__ gates) {
  const unsigned lane = threadIdx.x & 31;
  const unsigned wave = threadIdx.x >> 5;
  const unsigned job = blockIdx.x * 8 + wave;    // 0..127
  const unsigned mtile = job & 1;
  const unsigned ntile0 = (job >> 1) * 2;        // 0,2,..,126
  const unsigned arow = mtile * 16 + (lane & 15);
  const unsigned nc = lane & 15;
  const _Float16* b0 = wcat16 + (size_t)(ntile0 * 16 + nc) * KCAT;
  const _Float16* b1 = b0 + (size_t)16 * KCAT;
  v8f acc0 = {}, acc1 = {};
  {
    const _Float16* a0 = x16_t + (size_t)arow * 512;
    for (int kb = 0; kb < 16; ++kb) {
      int k0 = kb * 32;
      v16h a = load_frag16(a0 + k0, lane);
      acc0 = wmma_f16(a, load_frag16(b0 + k0, lane), acc0);
      acc1 = wmma_f16(a, load_frag16(b1 + k0, lane), acc1);
    }
  }
  {
    const _Float16* a0 = hhat16 + (size_t)arow * 512;
    for (int kb = 0; kb < 16; ++kb) {
      int k0 = kb * 32;
      v16h a = load_frag16(a0 + k0, lane);
      acc0 = wmma_f16(a, load_frag16(b0 + 512 + k0, lane), acc0);
      acc1 = wmma_f16(a, load_frag16(b1 + 512 + k0, lane), acc1);
    }
  }
  {
    const _Float16* a0 = h16 + (size_t)arow * 512;
    for (int kb = 0; kb < 16; ++kb) {
      int k0 = kb * 32;
      v16h a = load_frag16(a0 + k0, lane);
      acc0 = wmma_f16(a, load_frag16(b0 + 1024 + k0, lane), acc0);
      acc1 = wmma_f16(a, load_frag16(b1 + 1024 + k0, lane), acc1);
    }
  }
  const unsigned mb = mtile * 16 + ((lane & 16) ? 8 : 0);
#pragma unroll
  for (int r = 0; r < 8; ++r) {
    gates[(size_t)(mb + r) * NG + ntile0 * 16 + nc] = acc0[r];
    gates[(size_t)(mb + r) * NG + (ntile0 + 1) * 16 + nc] = acc1[r];
  }
}

__global__ void lstm_cell_kernel(const float* __restrict__ gates,
                                 const float* __restrict__ b_ih,
                                 const float* __restrict__ b_hh,
                                 float* __restrict__ h, float* __restrict__ c,
                                 _Float16* __restrict__ h16) {
  int i = blockIdx.x * 256 + threadIdx.x;
  if (i >= BB * HH) return;
  int b = i >> 9, j = i & 511;
  const float* g = gates + (size_t)b * NG;
  float ig = g[j]        + b_ih[j]        + b_hh[j];
  float fg = g[512 + j]  + b_ih[512 + j]  + b_hh[512 + j];
  float gg = g[1024 + j] + b_ih[1024 + j] + b_hh[1024 + j];
  float og = g[1536 + j] + b_ih[1536 + j] + b_hh[1536 + j];
  ig = 1.f / (1.f + expf(-ig));
  fg = 1.f / (1.f + expf(-fg));
  gg = tanhf(gg);
  og = 1.f / (1.f + expf(-og));
  float cn = fg * c[i] + ig * gg;
  float hn = og * tanhf(cn);
  c[i] = cn; h[i] = hn; h16[i] = (_Float16)hn;
}

// key_p = h @ Wp^T + bp   (M=32, N=1024, K=512). 2 N-tiles per wave -> 64 jobs.
__global__ void keyp_gemm_kernel(const _Float16* __restrict__ h16,
                                 const _Float16* __restrict__ wp16,
                                 const float* __restrict__ bp,
                                 float* __restrict__ keyp) {
  const unsigned lane = threadIdx.x & 31;
  const unsigned wave = threadIdx.x >> 5;
  const unsigned job = blockIdx.x * 8 + wave;    // 0..63
  const unsigned mtile = job & 1;
  const unsigned ntile0 = (job >> 1) * 2;        // 0,2,..,62
  const unsigned arow = mtile * 16 + (lane & 15);
  const unsigned nc = lane & 15;
  const _Float16* a0 = h16 + (size_t)arow * 512;
  const _Float16* b0 = wp16 + (size_t)(ntile0 * 16 + nc) * 512;
  const _Float16* b1 = b0 + (size_t)16 * 512;
  v8f acc0 = {}, acc1 = {};
  for (int kb = 0; kb < 16; ++kb) {
    int k0 = kb * 32;
    v16h a = load_frag16(a0 + k0, lane);
    acc0 = wmma_f16(a, load_frag16(b0 + k0, lane), acc0);
    acc1 = wmma_f16(a, load_frag16(b1 + k0, lane), acc1);
  }
  const unsigned mb = mtile * 16 + ((lane & 16) ? 8 : 0);
  float bias0 = bp[ntile0 * 16 + nc];
  float bias1 = bp[(ntile0 + 1) * 16 + nc];
#pragma unroll
  for (int r = 0; r < 8; ++r) {
    keyp[(size_t)(mb + r) * SDD + ntile0 * 16 + nc] = acc0[r] + bias0;
    keyp[(size_t)(mb + r) * SDD + (ntile0 + 1) * 16 + nc] = acc1[r] + bias1;
  }
}

// attention: scores -> softmax -> context (per batch b), ct16 out (32x1024 f16)
__global__ void attn_kernel(const float* __restrict__ src,
                            const float* __restrict__ keyp,
                            _Float16* __restrict__ ct16) {
  const int b = blockIdx.x;
  const int tid = threadIdx.x;
  const int lane = tid & 31, wid = tid >> 5;   // 8 waves
  __shared__ float sc[SS];
  __shared__ float pr[SS];
  __shared__ float sinv;
  const float* srcb = src + (size_t)b * SS * SDD;
  const float* kp = keyp + (size_t)b * SDD;
  const float scale = 0.044194173824159216f;   // 1/sqrt(512)
  for (int s = wid * 8; s < wid * 8 + 8; ++s) {
    float part = 0.f;
    for (int d = lane; d < SDD; d += 32)
      part += kp[d] * srcb[(size_t)s * SDD + d];
#pragma unroll
    for (int off = 16; off; off >>= 1)
      part += __shfl_xor(part, off, 32);
    if (lane == 0) sc[s] = part * scale;
  }
  __syncthreads();
  if (tid == 0) {
    float mx = sc[0];
    for (int s = 1; s < SS; ++s) mx = fmaxf(mx, sc[s]);
    float sum = 0.f;
    for (int s = 0; s < SS; ++s) { float e = expf(sc[s] - mx); pr[s] = e; sum += e; }
    sinv = 1.f / sum;
  }
  __syncthreads();
  const float inv = sinv;
  for (int d = tid; d < SDD; d += 256) {
    float a = 0.f;
    for (int s = 0; s < SS; ++s) a += pr[s] * srcb[(size_t)s * SDD + d];
    ct16[(size_t)b * SDD + d] = (_Float16)(a * inv);
  }
}

// h_hat = tanh([h, c_t] @ Wah^T + bah)  (M=32, N=512, K=1536).
// 2 N-tiles per wave -> 32 jobs; segment-split K loops (global loads only).
__global__ void hhat_gemm_kernel(const _Float16* __restrict__ h16,
                                 const _Float16* __restrict__ ct16,
                                 const _Float16* __restrict__ wah16,
                                 const float* __restrict__ bah,
                                 float* __restrict__ hhat_f32,
                                 _Float16* __restrict__ hhat16,
                                 _Float16* __restrict__ outs16_t) {
  const unsigned lane = threadIdx.x & 31;
  const unsigned wave = threadIdx.x >> 5;
  const unsigned job = blockIdx.x * 8 + wave;    // 0..31
  const unsigned mtile = job & 1;
  const unsigned ntile0 = (job >> 1) * 2;        // 0,2,..,30
  const unsigned arow = mtile * 16 + (lane & 15);
  const unsigned nc = lane & 15;
  const _Float16* b0 = wah16 + (size_t)(ntile0 * 16 + nc) * KCAT;
  const _Float16* b1 = b0 + (size_t)16 * KCAT;
  v8f acc0 = {}, acc1 = {};
  {
    const _Float16* a0 = h16 + (size_t)arow * 512;
    for (int kb = 0; kb < 16; ++kb) {
      int k0 = kb * 32;
      v16h a = load_frag16(a0 + k0, lane);
      acc0 = wmma_f16(a, load_frag16(b0 + k0, lane), acc0);
      acc1 = wmma_f16(a, load_frag16(b1 + k0, lane), acc1);
    }
  }
  {
    const _Float16* a0 = ct16 + (size_t)arow * 1024;
    for (int kb = 0; kb < 32; ++kb) {
      int k0 = kb * 32;
      v16h a = load_frag16(a0 + k0, lane);
      acc0 = wmma_f16(a, load_frag16(b0 + 512 + k0, lane), acc0);
      acc1 = wmma_f16(a, load_frag16(b1 + 512 + k0, lane), acc1);
    }
  }
  const unsigned mb = mtile * 16 + ((lane & 16) ? 8 : 0);
  float bias0 = bah[ntile0 * 16 + nc];
  float bias1 = bah[(ntile0 + 1) * 16 + nc];
#pragma unroll
  for (int r = 0; r < 8; ++r) {
    float v0 = tanhf(acc0[r] + bias0);
    float v1 = tanhf(acc1[r] + bias1);
    size_t i0 = (size_t)(mb + r) * HH + ntile0 * 16 + nc;
    size_t i1 = (size_t)(mb + r) * HH + (ntile0 + 1) * 16 + nc;
    hhat_f32[i0] = v0; hhat_f32[i1] = v1;
    _Float16 h0 = (_Float16)v0, h1 = (_Float16)v1;
    hhat16[i0] = h0; hhat16[i1] = h1;
    outs16_t[i0] = h0; outs16_t[i1] = h1;
  }
}

// ---------------- final vocab projection ----------------
// out[b][t][v] = outs16[t*32+b][:] . Wfc16[v][:] + bfc[v]
// M = 2048 (row m = t*32+b), N = 32000, K = 512.
// Each wave: 16x64 output (4 N-tiles share one A fragment per K-step).
__global__ void fc_gemm_kernel(const _Float16* __restrict__ outs16,
                               const _Float16* __restrict__ wfc16,
                               const float* __restrict__ bfc,
                               float* __restrict__ out) {
  const unsigned lane = threadIdx.x & 31;
  const unsigned wave = threadIdx.x >> 5;
  const unsigned job = blockIdx.x * 8 + wave;    // 0..503 (500 used)
  const unsigned ntile0 = job * 4;               // 4 N-tiles per wave
  if (ntile0 >= VV / 16) return;                 // wave-uniform exit
  const unsigned mtile = blockIdx.y;             // 0..127
  const unsigned arow = mtile * 16 + (lane & 15);
  const unsigned nc = lane & 15;
  const _Float16* aptr = outs16 + (size_t)arow * HH;
  const _Float16* b0 = wfc16 + (size_t)(ntile0 * 16 + nc) * HH;
  const _Float16* b1 = b0 + (size_t)16 * HH;
  const _Float16* b2 = b0 + (size_t)32 * HH;
  const _Float16* b3 = b0 + (size_t)48 * HH;
  v8f acc[4] = {{}, {}, {}, {}};
  for (int kb = 0; kb < 16; ++kb) {
    int k0 = kb * 32;
    // stream Wfc: prefetch 4 K-blocks (256B) ahead -> global_prefetch_b8
    __builtin_prefetch(b0 + k0 + 128, 0, 1);
    __builtin_prefetch(b1 + k0 + 128, 0, 1);
    __builtin_prefetch(b2 + k0 + 128, 0, 1);
    __builtin_prefetch(b3 + k0 + 128, 0, 1);
    v16h a = load_frag16(aptr + k0, lane);
    acc[0] = wmma_f16(a, load_frag16(b0 + k0, lane), acc[0]);
    acc[1] = wmma_f16(a, load_frag16(b1 + k0, lane), acc[1]);
    acc[2] = wmma_f16(a, load_frag16(b2 + k0, lane), acc[2]);
    acc[3] = wmma_f16(a, load_frag16(b3 + k0, lane), acc[3]);
  }
  const unsigned mb = mtile * 16 + ((lane & 16) ? 8 : 0);
#pragma unroll
  for (int j = 0; j < 4; ++j) {
    unsigned ncol = (ntile0 + j) * 16 + nc;
    float bias = bfc[ncol];
#pragma unroll
    for (int r = 0; r < 8; ++r) {
      unsigned m = mb + r;
      unsigned b = m & 31, t = m >> 5;
      out[((size_t)b * TT + t) * VV + ncol] = acc[j][r] + bias;
    }
  }
}

__global__ void copy_tail_kernel(const float* __restrict__ h,
                                 const float* __restrict__ c,
                                 const float* __restrict__ hhat,
                                 float* __restrict__ out) {
  int i = blockIdx.x * 256 + threadIdx.x;
  if (i >= BB * HH) return;
  float* base = out + (size_t)BB * TT * VV;
  base[i] = h[i];
  base[BB * HH + i] = c[i];
  base[2 * BB * HH + i] = hhat[i];
}

// ---------------- host launcher ----------------
extern "C" void kernel_launch(void* const* d_in, const int* in_sizes, int n_in,
                              void* d_out, int out_size, void* d_ws, size_t ws_size,
                              hipStream_t stream) {
  (void)in_sizes; (void)n_in; (void)out_size; (void)ws_size;
  const float* src   = (const float*)d_in[0];
  const int*   trg   = (const int*)  d_in[1];
  const float* emb   = (const float*)d_in[2];
  const float* W_ih  = (const float*)d_in[3];
  const float* b_ih  = (const float*)d_in[4];
  const float* W_hh  = (const float*)d_in[5];
  const float* b_hh  = (const float*)d_in[6];
  const float* Wp    = (const float*)d_in[7];
  const float* bp    = (const float*)d_in[8];
  const float* Wah   = (const float*)d_in[9];
  const float* bah   = (const float*)d_in[10];
  const float* Wfc   = (const float*)d_in[11];
  const float* bfc   = (const float*)d_in[12];
  const float* iff   = (const float*)d_in[13];
  const float* hinit = (const float*)d_in[14];
  float* out = (float*)d_out;

  // workspace carve-up (256B aligned regions)
  char* ws = (char*)d_ws;
  size_t off = 0;
  auto take = [&](size_t bytes) -> char* {
    char* p = ws + off;
    off = (off + bytes + 255) & ~(size_t)255;
    return p;
  };
  _Float16* wcat16 = (_Float16*)take((size_t)NG * KCAT * 2);      // 6.0 MB
  _Float16* wp16   = (_Float16*)take((size_t)SDD * HH * 2);       // 1.0 MB
  _Float16* wah16  = (_Float16*)take((size_t)HH * KCAT * 2);      // 1.5 MB
  _Float16* wfc16  = (_Float16*)take((size_t)VV * HH * 2);        // 32.8 MB
  _Float16* x16    = (_Float16*)take((size_t)TT * BB * EE * 2);   // 2.0 MB
  _Float16* outs16 = (_Float16*)take((size_t)TT * BB * HH * 2);   // 2.0 MB
  _Float16* h16    = (_Float16*)take((size_t)BB * HH * 2);
  _Float16* hhat16 = (_Float16*)take((size_t)BB * HH * 2);
  _Float16* ct16   = (_Float16*)take((size_t)BB * SDD * 2);
  float* hf    = (float*)take((size_t)BB * HH * 4);
  float* cf    = (float*)take((size_t)BB * HH * 4);
  float* hhatf = (float*)take((size_t)BB * HH * 4);
  float* gates = (float*)take((size_t)BB * NG * 4);
  float* keyp  = (float*)take((size_t)BB * SDD * 4);

  // ---- one-time setup ----
  build_wcat_kernel<<<(NG * KCAT + 255) / 256, 256, 0, stream>>>(W_ih, W_hh, wcat16);
  cvt_f16_kernel<<<(SDD * HH + 255) / 256, 256, 0, stream>>>(Wp, wp16, SDD * HH);
  cvt_f16_kernel<<<(HH * KCAT + 255) / 256, 256, 0, stream>>>(Wah, wah16, HH * KCAT);
  cvt_f16_kernel<<<(VV * HH + 255) / 256, 256, 0, stream>>>(Wfc, wfc16, VV * HH);
  embed_kernel<<<(TT * BB * EE + 255) / 256, 256, 0, stream>>>(emb, trg, x16);
  init_state_kernel<<<(BB * HH + 255) / 256, 256, 0, stream>>>(hinit, iff, hf, cf,
                                                               hhatf, h16, hhat16);
  // ---- recurrence ----
  for (int t = 0; t < TT; ++t) {
    const _Float16* x16_t = x16 + (size_t)t * BB * EE;
    _Float16* outs16_t = outs16 + (size_t)t * BB * HH;
    gate_gemm_kernel<<<16, 256, 0, stream>>>(x16_t, hhat16, h16, wcat16, gates);
    lstm_cell_kernel<<<(BB * HH + 255) / 256, 256, 0, stream>>>(gates, b_ih, b_hh,
                                                                hf, cf, h16);
    keyp_gemm_kernel<<<8, 256, 0, stream>>>(h16, wp16, bp, keyp);
    attn_kernel<<<BB, 256, 0, stream>>>(src, keyp, ct16);
    hhat_gemm_kernel<<<4, 256, 0, stream>>>(h16, ct16, wah16, bah,
                                            hhatf, hhat16, outs16_t);
  }
  // ---- vocab projection: 128 M-tiles x 500 wave-jobs (16x64 per wave) ----
  fc_gemm_kernel<<<dim3(63, (BB * TT) / 16), 256, 0, stream>>>(
      outs16, wfc16, bfc, out);
  copy_tail_kernel<<<(BB * HH + 255) / 256, 256, 0, stream>>>(hf, cf, hhatf, out);
}
